// SelfAttention1d_68624987456221
// MI455X (gfx1250) — compile-verified, matrix operand order checked by reading, not run on hardware
//
#include <hip/hip_runtime.h>

typedef __attribute__((ext_vector_type(16))) _Float16 v16h;
typedef __attribute__((ext_vector_type(8)))  float    v8f;
typedef unsigned int u32x4 __attribute__((ext_vector_type(4)));
typedef int          i32x8 __attribute__((ext_vector_type(8)));
typedef int          i32x4 __attribute__((ext_vector_type(4)));

constexpr int CDIM  = 256;
constexpr int HEADS = 8;
constexpr int DHEAD = 32;
constexpr int SEQ   = 2048;
constexpr int BATCH = 4;
constexpr int ROWS  = BATCH * SEQ;     // 8192
constexpr int NQKV  = 3 * CDIM;        // 768
constexpr float SCALE = 0.17677669529663687f; // 1/sqrt(32)

// CDNA5 16-bit A/B fragment K-index mapping (ISA 7.12.2):
// lane<16: K = {0..7, 16..23}; lane>=16: K = {8..15, 24..31}
__device__ __forceinline__ int kmap(int lane, int i) {
  int koff = (lane & 16) ? 8 : 0;
  return (i < 8) ? (koff + i) : (16 + koff + (i - 8));
}

__device__ __forceinline__ v8f wmma_f16(v16h a, v16h b, v8f c) {
  return __builtin_amdgcn_wmma_f32_16x16x32_f16(false, a, false, b, (short)0, c,
                                                false, false);
}

// TDM: 1-D contiguous block copy global -> LDS, data_size = 8 bytes.
// D# group0: count=1 | lds_addr | global_addr(57b) | type=2.
// D# group1: data_size=3(8B); tensor_dim0=tile_dim0=elems8; dims1.. = 1.
__device__ __forceinline__ void tdm_load_1d(unsigned ldsAddr, const void* gaddr,
                                            unsigned elems8) {
  unsigned long long ga = (unsigned long long)gaddr;
  u32x4 g0;
  g0[0] = 1u;                                              // count=1
  g0[1] = ldsAddr;                                         // lds_addr
  g0[2] = (unsigned)(ga & 0xFFFFFFFFu);                    // global_addr[31:0]
  g0[3] = (unsigned)((ga >> 32) & 0x01FFFFFFu) | (2u << 30); // [56:32] | type=2
  i32x8 g1;
  g1[0] = 3 << 16;                                         // data_size = 8B
  g1[1] = (int)(elems8 << 16);                             // tensor_dim0[15:0]
  g1[2] = (int)((elems8 >> 16) & 0xFFFFu) | (1 << 16);     // dim0[31:16] | tensor_dim1=1
  g1[3] = (int)(elems8 << 16);                             // tile_dim0 (<=65535)
  g1[4] = 1;                                               // tile_dim1=1
  g1[5] = (int)elems8;                                     // tensor_dim0_stride[31:0]
  g1[6] = 0;
  g1[7] = 0;
  i32x4 z4 = {0, 0, 0, 0};
  i32x8 z8 = {0, 0, 0, 0, 0, 0, 0, 0};
  __builtin_amdgcn_tensor_load_to_lds(g0, g1, z4, z4, z8, 0);
}

// ---------------------------------------------------------------------------
// Kernel 1: qkv = x @ w_qkv^T + b_qkv, scattered to head-major f16 Q/K/V.
// Q is pre-scaled by 1/sqrt(Dh).
// grid: (ROWS/16, NQKV/128), block 256 (8 waves; wave w owns N-tile w).
// ---------------------------------------------------------------------------
__global__ __launch_bounds__(256)
void qkv_kernel(const float* __restrict__ x, const float* __restrict__ w,
                const float* __restrict__ bqkv,
                _Float16* __restrict__ qf, _Float16* __restrict__ kf,
                _Float16* __restrict__ vf) {
  constexpr int AST = CDIM + 8;
  __shared__ _Float16 sA[16 * AST];
  const int tid = threadIdx.x;
  const int m0  = blockIdx.x * 16;
  for (int e = tid; e < 16 * CDIM; e += 256) {
    int r = e >> 8, c = e & (CDIM - 1);
    sA[r * AST + c] = (_Float16)x[(size_t)(m0 + r) * CDIM + c];
  }
  __syncthreads();
  const int wave = tid >> 5, lane = tid & 31;
  const int ncol = blockIdx.y * 128 + wave * 16 + (lane & 15);
  v8f acc = {};
#pragma unroll
  for (int k0 = 0; k0 < CDIM; k0 += 32) {
    v16h a, b;
#pragma unroll
    for (int i = 0; i < 16; ++i) {
      int k = k0 + kmap(lane, i);
      a[i] = sA[(lane & 15) * AST + k];
      b[i] = (_Float16)w[(size_t)ncol * CDIM + k];  // B[k][n] = w[n][k]
    }
    acc = wmma_f16(a, b, acc);
  }
  const int h    = ncol / (3 * DHEAD);
  const int rem  = ncol % (3 * DHEAD);
  const int comp = rem / DHEAD;
  const int d    = rem % DHEAD;
  const float bias = bqkv[ncol];
#pragma unroll
  for (int r = 0; r < 8; ++r) {
    int row = m0 + ((lane >> 4) << 3) + r;
    int bb = row / SEQ, ll = row % SEQ;
    float val = acc[r] + bias;
    size_t idx = (((size_t)(bb * HEADS + h) * SEQ) + ll) * DHEAD + d;
    if (comp == 0)      qf[idx] = (_Float16)(val * SCALE);
    else if (comp == 1) kf[idx] = (_Float16)val;
    else                vf[idx] = (_Float16)val;
  }
}

// ---------------------------------------------------------------------------
// Kernel 2: flash attention. One (b,h) per blockIdx.x; 256 query rows per
// blockIdx.y. Full K,V for the head live in LDS (256 KB, CDNA5 320KB WGP LDS).
// K staged via TENSOR_LOAD_TO_LDS (TDM, TENSORcnt); V staged via
// GLOBAL_LOAD_ASYNC_TO_LDS_B128 (ASYNCcnt). Online softmax; P repacked
// C-layout -> A-layout via per-wave LDS slot.
// ---------------------------------------------------------------------------
__global__ __launch_bounds__(256)
void attn_kernel(const _Float16* __restrict__ qf, const _Float16* __restrict__ kf,
                 const _Float16* __restrict__ vf, _Float16* __restrict__ aout) {
  extern __shared__ char smem[];
  _Float16* sK = (_Float16*)smem;          // SEQ*DHEAD halves
  _Float16* sV = sK + SEQ * DHEAD;         // SEQ*DHEAD halves
  _Float16* sP = sV + SEQ * DHEAD;         // 8 waves * 16*32 halves
  const int tid = threadIdx.x;
  const int bh  = blockIdx.x;
  const int bb  = bh / HEADS, hh = bh % HEADS;
  const size_t base = (size_t)bh * SEQ * DHEAD;
  const int wave = tid >> 5, lane = tid & 31;

  {
    const unsigned ldsK = (unsigned)(size_t)(void*)sK;
    const unsigned ldsV = (unsigned)(size_t)(void*)sV;
    const unsigned long long gV = (unsigned long long)(const void*)(vf + base);
    // V: 8192 16-byte chunks via per-lane async loads (ASYNCcnt path).
    for (int e = tid; e < SEQ * DHEAD / 8; e += 256) {
      unsigned loff = (unsigned)e * 16u;
      asm volatile("global_load_async_to_lds_b128 %0, %1, off"
                   :: "v"(ldsV + loff), "v"(gV + loff) : "memory");
    }
    // K: one TDM descriptor moves the whole 128 KB tile (TENSORcnt path).
    if (wave == 0) {
      tdm_load_1d(ldsK, kf + base, SEQ * DHEAD / 4);
      __builtin_amdgcn_s_wait_tensorcnt(0);
    }
    asm volatile("s_wait_asynccnt 0x0" ::: "memory");
  }
  __syncthreads();

  _Float16* myP = sP + wave * 512;

  for (int t = 0; t < 2; ++t) {
    const int qrow0 = blockIdx.y * 256 + (wave * 2 + t) * 16;
    v16h qa;
#pragma unroll
    for (int i = 0; i < 16; ++i)
      qa[i] = qf[base + (size_t)(qrow0 + (lane & 15)) * DHEAD + kmap(lane, i)];

    v8f o0 = {}, o1 = {};
    float mrun[8], lrun[8];
#pragma unroll
    for (int r = 0; r < 8; ++r) { mrun[r] = -1e30f; lrun[r] = 0.f; }

#pragma unroll 2
    for (int j0 = 0; j0 < SEQ; j0 += 32) {
      v16h kb0, kb1;
#pragma unroll
      for (int i = 0; i < 16; ++i) {
        int k = kmap(lane, i);
        kb0[i] = sK[(j0 + (lane & 15)) * DHEAD + k];        // B[d][j] = K[j][d]
        kb1[i] = sK[(j0 + 16 + (lane & 15)) * DHEAD + k];
      }
      v8f z = {};
      v8f s0 = wmma_f16(qa, kb0, z);
      v8f s1 = wmma_f16(qa, kb1, z);

#pragma unroll
      for (int r = 0; r < 8; ++r) {
        float tm = fmaxf(s0[r], s1[r]);
#pragma unroll
        for (int m = 8; m >= 1; m >>= 1)
          tm = fmaxf(tm, __shfl_xor(tm, m, 16));
        float newm = fmaxf(mrun[r], tm);
        float corr = __expf(mrun[r] - newm);
        float p0 = __expf(s0[r] - newm);
        float p1 = __expf(s1[r] - newm);
        float rs = p0 + p1;
#pragma unroll
        for (int m = 8; m >= 1; m >>= 1)
          rs += __shfl_xor(rs, m, 16);
        lrun[r] = lrun[r] * corr + rs;
        mrun[r] = newm;
        o0[r] *= corr;
        o1[r] *= corr;
        int prow = ((lane >> 4) << 3) + r;               // C-frag row
        myP[prow * 32 + (lane & 15)]      = (_Float16)p0;
        myP[prow * 32 + 16 + (lane & 15)] = (_Float16)p1;
      }
      v16h pa, vb0, vb1;
#pragma unroll
      for (int i = 0; i < 16; ++i) {
        int k = kmap(lane, i);
        pa[i]  = myP[(lane & 15) * 32 + k];              // A-frag of P
        vb0[i] = sV[(j0 + k) * DHEAD + (lane & 15)];     // B[k][d]
        vb1[i] = sV[(j0 + k) * DHEAD + 16 + (lane & 15)];
      }
      o0 = wmma_f16(pa, vb0, o0);
      o1 = wmma_f16(pa, vb1, o1);
    }

#pragma unroll
    for (int r = 0; r < 8; ++r) {
      float inv = 1.0f / lrun[r];
      int qrow = qrow0 + ((lane >> 4) << 3) + r;
      size_t orow = ((size_t)bb * SEQ + qrow) * CDIM + hh * DHEAD;
      aout[orow + (lane & 15)]      = (_Float16)(o0[r] * inv);
      aout[orow + 16 + (lane & 15)] = (_Float16)(o1[r] * inv);
    }
  }
}

// ---------------------------------------------------------------------------
// Kernel 3: out = aout(f16) @ w_proj^T + b_proj  (fp32 result)
// grid: (ROWS/16, CDIM/128), block 256.
// ---------------------------------------------------------------------------
__global__ __launch_bounds__(256)
void proj_kernel(const _Float16* __restrict__ aout, const float* __restrict__ w,
                 const float* __restrict__ bias, float* __restrict__ out) {
  constexpr int AST = CDIM + 8;
  __shared__ _Float16 sA[16 * AST];
  const int tid = threadIdx.x;
  const int m0  = blockIdx.x * 16;
  for (int e = tid; e < 16 * CDIM; e += 256) {
    int r = e >> 8, c = e & (CDIM - 1);
    sA[r * AST + c] = aout[(size_t)(m0 + r) * CDIM + c];
  }
  __syncthreads();
  const int wave = tid >> 5, lane = tid & 31;
  const int ncol = blockIdx.y * 128 + wave * 16 + (lane & 15);
  v8f acc = {};
#pragma unroll
  for (int k0 = 0; k0 < CDIM; k0 += 32) {
    v16h a, b;
#pragma unroll
    for (int i = 0; i < 16; ++i) {
      int k = k0 + kmap(lane, i);
      a[i] = sA[(lane & 15) * AST + k];
      b[i] = (_Float16)w[(size_t)ncol * CDIM + k];
    }
    acc = wmma_f16(a, b, acc);
  }
  const float bv = bias[ncol];
#pragma unroll
  for (int r = 0; r < 8; ++r) {
    int row = m0 + ((lane >> 4) << 3) + r;
    out[(size_t)row * CDIM + ncol] = acc[r] + bv;
  }
}

// ---------------------------------------------------------------------------
extern "C" void kernel_launch(void* const* d_in, const int* in_sizes, int n_in,
                              void* d_out, int out_size, void* d_ws, size_t ws_size,
                              hipStream_t stream) {
  const float* x      = (const float*)d_in[0];
  const float* w_qkv  = (const float*)d_in[1];
  const float* b_qkv  = (const float*)d_in[2];
  const float* w_proj = (const float*)d_in[3];
  const float* b_proj = (const float*)d_in[4];
  float* out = (float*)d_out;

  const size_t headElems = (size_t)BATCH * HEADS * SEQ * DHEAD; // 2M halves
  _Float16* qf   = (_Float16*)d_ws;
  _Float16* kf   = qf + headElems;
  _Float16* vf   = kf + headElems;
  _Float16* aout = vf + headElems;   // ROWS*CDIM halves; total ws = 16 MB

  qkv_kernel<<<dim3(ROWS / 16, NQKV / 128), 256, 0, stream>>>(
      x, w_qkv, b_qkv, qf, kf, vf);

  size_t smem2 = (size_t)(2 * SEQ * DHEAD + 8 * 16 * 32) * sizeof(_Float16);
  attn_kernel<<<dim3(BATCH * HEADS, SEQ / 256), 256, smem2, stream>>>(
      qf, kf, vf, aout);

  proj_kernel<<<dim3(ROWS / 16, CDIM / 128), 256, 0, stream>>>(
      aout, w_proj, b_proj, out);
}
